// NeuralTransformationCache_55044300866028
// MI455X (gfx1250) — matrix operand order repaired and checked by reading, not compiled, not run on hardware
//
#include <hip/hip_runtime.h>

// ---------------------------------------------------------------------------
// NeuralTransformationCache: hash-grid encode + fused 64-64-64-8 MLP (WMMA)
// Target: gfx1250 (CDNA5, wave32, v_wmma_f32_16x16x32_f16)
// LDS rows padded to 144B (activations/weights) and 80B (resi) to rotate
// across the 64x4B LDS banks (256B conflict-free span).
// ---------------------------------------------------------------------------

typedef __attribute__((ext_vector_type(16))) _Float16 v16h;
typedef __attribute__((ext_vector_type(8)))  _Float16 v8h;
typedef __attribute__((ext_vector_type(8)))  float    v8f;

#define L_LEVELS  16
#define T_SIZE    32768
#define BLOCK_PTS 128
#define THREADS   256
#define ASTRIDE   72   // padded row stride (halves) for activations/weights
#define RSTRIDE   20   // padded row stride (floats) for resi

// --- WMMA fragment loaders (layouts per CDNA5 ISA §7.12.2) ------

// A: 16x32 f16 (MxK). lane<16: M=lane&15, K-base 0; lane>=16: K-base 8.
// halves 0..7 = K kb..kb+7, halves 8..15 = K kb+16..kb+23. Two 16B LDS reads.
__device__ __forceinline__ v16h load_a_frag(const _Float16* base, int rowBase,
                                            int kc, int lane) {
    const int m  = lane & 15;
    const int kb = (lane < 16) ? 0 : 8;
    const _Float16* rp = base + (rowBase + m) * ASTRIDE + kc * 32 + kb;
    union { v16h v; v8h h[2]; } u;
    u.h[0] = *(const v8h*)(rp);        // K = kb .. kb+7
    u.h[1] = *(const v8h*)(rp + 16);   // K = kb+16 .. kb+23
    return u.v;
}

// B: 32x16 f16 (KxN), weights staged transposed in LDS as wT[n][k] so the
// K-halves for one lane are in two contiguous runs. lanes 0-15: K=0..7,16..23,
// lanes 16-31: K=8..15,24..31, N = lane&15. Two 16B LDS reads.
__device__ __forceinline__ v16h load_b_frag(const _Float16* wT, int kc, int lane) {
    const int n  = lane & 15;
    const int ko = (lane < 16) ? 0 : 8;
    const _Float16* rp = wT + n * ASTRIDE + kc * 32 + ko;
    union { v16h v; v8h h[2]; } u;
    u.h[0] = *(const v8h*)(rp);
    u.h[1] = *(const v8h*)(rp + 16);
    return u.v;
}

__global__ __launch_bounds__(THREADS)
void NeuralTransformationCache_55044300866028_kernel(
        const float* __restrict__ xyz,
        const float* __restrict__ bmin,
        const float* __restrict__ bmax,
        const float* __restrict__ table,   // [L][T][4] f32
        const float* __restrict__ w0,      // [64][64]
        const float* __restrict__ w1,      // [64][64]
        const float* __restrict__ w2,      // [64][8]
        float* __restrict__ out,           // [N mask | 3N d_xyz | 4N d_rot]
        int N)
{
    __shared__ _Float16 encLDS[BLOCK_PTS][ASTRIDE]; // activations (reused as h2)
    __shared__ _Float16 hLDS  [BLOCK_PTS][ASTRIDE]; // hidden layer 1
    __shared__ _Float16 w0T[64][ASTRIDE];           // [out][in] f16
    __shared__ _Float16 w1T[64][ASTRIDE];
    __shared__ _Float16 w2T[16][ASTRIDE];           // padded N: rows 8..15 zero
    __shared__ float    resi[BLOCK_PTS][RSTRIDE];

    const int t    = threadIdx.x;
    const int lane = t & 31;
    const int wave = t >> 5;
    const int blockBase = blockIdx.x * BLOCK_PTS;

    // ---- stage 0: stage weights into LDS, f16, transposed to [out][in] ----
    for (int i = t; i < 64 * 64; i += THREADS) {
        const int k = i >> 6, n = i & 63;     // global layout w[k][n]
        w0T[n][k] = (_Float16)w0[i];
        w1T[n][k] = (_Float16)w1[i];
    }
    for (int i = t; i < 16 * 64; i += THREADS) {
        const int n = i >> 6, k = i & 63;
        w2T[n][k] = (n < 8) ? (_Float16)w2[k * 8 + n] : (_Float16)0.0f;
    }

    // ---- stage 1: hash-grid encode (2 threads/point, 8 levels each) -------
    {
        const int pid  = t & (BLOCK_PTS - 1);
        const int half = t >> 7;              // 0 or 1
        int p = blockBase + pid;
        if (p >= N) p = (N > 0) ? N - 1 : 0;  // clamp tail; no divergence/return

        const float bx0 = bmin[0], by0 = bmin[1], bz0 = bmin[2];
        const float bx1 = bmax[0], by1 = bmax[1], bz1 = bmax[2];
        float cx = (xyz[3 * p + 0] - bx0) / (bx1 - bx0);
        float cy = (xyz[3 * p + 1] - by0) / (by1 - by0);
        float cz = (xyz[3 * p + 2] - bz0) / (bz1 - bz0);
        cx = fminf(fmaxf(cx, 0.0f), 1.0f);
        cy = fminf(fmaxf(cy, 0.0f), 1.0f);
        cz = fminf(fmaxf(cz, 0.0f), 1.0f);

        const int l0 = half * 8;
        for (int l = l0; l < l0 + 8; ++l) {
            const float scale = (float)(1u << (l + 4));  // 16 * 2^l
            const float px = cx * scale, py = cy * scale, pz = cz * scale;
            const float fx = floorf(px), fy = floorf(py), fz = floorf(pz);
            const float rx = px - fx, ry = py - fy, rz = pz - fz;
            const unsigned ix = (unsigned)fx, iy = (unsigned)fy, iz = (unsigned)fz;

            const float4* tb = (const float4*)table + (size_t)l * T_SIZE;
            float a0 = 0.f, a1 = 0.f, a2 = 0.f, a3 = 0.f;
            #pragma unroll
            for (int c = 0; c < 8; ++c) {
                const unsigned ox = (c >> 2) & 1u, oy = (c >> 1) & 1u, oz = c & 1u;
                const unsigned h0 = ix + ox;                         // prime 1
                const unsigned h1 = (iy + oy) * 2654435761u;
                const unsigned h2 = (iz + oz) * 805459861u;
                const unsigned idx = (h0 ^ h1 ^ h2) & (T_SIZE - 1);
                const float4 f = tb[idx];                            // L2-resident gather
                const float w = (ox ? rx : 1.0f - rx) *
                                (oy ? ry : 1.0f - ry) *
                                (oz ? rz : 1.0f - rz);
                a0 += w * f.x; a1 += w * f.y; a2 += w * f.z; a3 += w * f.w;
            }
            encLDS[pid][l * 4 + 0] = (_Float16)a0;
            encLDS[pid][l * 4 + 1] = (_Float16)a1;
            encLDS[pid][l * 4 + 2] = (_Float16)a2;
            encLDS[pid][l * 4 + 3] = (_Float16)a3;
        }
    }
    __syncthreads();

    // ---- stage 2: layer 0  h = relu(enc @ w0)  -> hLDS --------------------
    // each wave owns one 16-row tile; 4 col tiles; K=64 = 2 wmma steps
    {
        const int rowBase = wave * 16;
        const int n  = lane & 15;
        const int mo = (lane < 16) ? 0 : 8;
        #pragma unroll
        for (int ct = 0; ct < 4; ++ct) {
            v8f acc = {};
            #pragma unroll
            for (int kc = 0; kc < 2; ++kc) {
                v16h a = load_a_frag(&encLDS[0][0], rowBase, kc, lane);
                v16h b = load_b_frag(&w0T[ct * 16][0], kc, lane);
                acc = __builtin_amdgcn_wmma_f32_16x16x32_f16(
                        false, a, false, b, (short)0, acc, false, false);
            }
            #pragma unroll
            for (int r = 0; r < 8; ++r) {
                const float v = acc[r] > 0.0f ? acc[r] : 0.0f;
                hLDS[rowBase + mo + r][ct * 16 + n] = (_Float16)v;
            }
        }
    }
    __syncthreads();

    // ---- stage 3: layer 1  h2 = relu(h @ w1)  -> encLDS (reuse) -----------
    {
        const int rowBase = wave * 16;
        const int n  = lane & 15;
        const int mo = (lane < 16) ? 0 : 8;
        #pragma unroll
        for (int ct = 0; ct < 4; ++ct) {
            v8f acc = {};
            #pragma unroll
            for (int kc = 0; kc < 2; ++kc) {
                v16h a = load_a_frag(&hLDS[0][0], rowBase, kc, lane);
                v16h b = load_b_frag(&w1T[ct * 16][0], kc, lane);
                acc = __builtin_amdgcn_wmma_f32_16x16x32_f16(
                        false, a, false, b, (short)0, acc, false, false);
            }
            #pragma unroll
            for (int r = 0; r < 8; ++r) {
                const float v = acc[r] > 0.0f ? acc[r] : 0.0f;
                encLDS[rowBase + mo + r][ct * 16 + n] = (_Float16)v;
            }
        }
    }
    __syncthreads();

    // ---- stage 4: layer 2  resi = h2 @ w2 (N padded 8->16) ----------------
    {
        const int rowBase = wave * 16;
        const int n  = lane & 15;
        const int mo = (lane < 16) ? 0 : 8;
        v8f acc = {};
        #pragma unroll
        for (int kc = 0; kc < 2; ++kc) {
            v16h a = load_a_frag(&encLDS[0][0], rowBase, kc, lane);
            v16h b = load_b_frag(&w2T[0][0], kc, lane);
            acc = __builtin_amdgcn_wmma_f32_16x16x32_f16(
                    false, a, false, b, (short)0, acc, false, false);
        }
        #pragma unroll
        for (int r = 0; r < 8; ++r)
            resi[rowBase + mo + r][n] = acc[r];
    }
    __syncthreads();

    // ---- stage 5: masked epilogue writes ----------------------------------
    if (t < BLOCK_PTS) {
        const int p = blockBase + t;
        if (p < N) {
            const float bx0 = bmin[0], by0 = bmin[1], bz0 = bmin[2];
            const float bx1 = bmax[0], by1 = bmax[1], bz1 = bmax[2];
            const float cx = (xyz[3 * p + 0] - bx0) / (bx1 - bx0);
            const float cy = (xyz[3 * p + 1] - by0) / (by1 - by0);
            const float cz = (xyz[3 * p + 2] - bz0) / (bz1 - bz0);
            const bool m = (cx >= 0.0f) & (cx <= 1.0f) &
                           (cy >= 0.0f) & (cy <= 1.0f) &
                           (cz >= 0.0f) & (cz <= 1.0f);
            out[p] = m ? 1.0f : 0.0f;                       // mask
            float* __restrict__ dxyz = out + (size_t)N;     // [N,3]
            float* __restrict__ drot = out + (size_t)4 * N; // [N,4]
            dxyz[3 * p + 0] = m ? resi[t][0] : 0.0f;
            dxyz[3 * p + 1] = m ? resi[t][1] : 0.0f;
            dxyz[3 * p + 2] = m ? resi[t][2] : 0.0f;
            drot[4 * p + 0] = m ? resi[t][3] : 1.0f;        // identity rot w
            drot[4 * p + 1] = m ? resi[t][4] : 0.0f;
            drot[4 * p + 2] = m ? resi[t][5] : 0.0f;
            drot[4 * p + 3] = m ? resi[t][6] : 0.0f;
        }
    }
}

extern "C" void kernel_launch(void* const* d_in, const int* in_sizes, int n_in,
                              void* d_out, int out_size, void* d_ws, size_t ws_size,
                              hipStream_t stream) {
    const float* xyz   = (const float*)d_in[0];
    const float* bmin  = (const float*)d_in[1];
    const float* bmax  = (const float*)d_in[2];
    const float* table = (const float*)d_in[3];
    const float* w0    = (const float*)d_in[4];
    const float* w1    = (const float*)d_in[5];
    const float* w2    = (const float*)d_in[6];
    float* out = (float*)d_out;

    const int N = in_sizes[0] / 3;
    if (N <= 0) return;
    const int blocks = (N + BLOCK_PTS - 1) / BLOCK_PTS;
    NeuralTransformationCache_55044300866028_kernel<<<blocks, THREADS, 0, stream>>>(
        xyz, bmin, bmax, table, w0, w1, w2, out, N);
}
